// AspectFocusedMetaLearning_46935402610853
// MI455X (gfx1250) — matrix operand order repaired and checked by reading, not compiled, not run on hardware
//
#include <hip/hip_runtime.h>

// ---------------------------------------------------------------------------
// Types for CDNA5 WMMA (wave32)
// ---------------------------------------------------------------------------
typedef __attribute__((ext_vector_type(16))) __bf16 v16bf;
typedef __attribute__((ext_vector_type(8)))  __bf16 v8bf;
typedef __attribute__((ext_vector_type(8)))  float  v8f;

#define HDIM 768
#define PER  24          // HDIM / 32 lanes

// GEMM tiling
#define BM 128
#define BN 128
#define BK 32
#define LDT 40           // padded K-stride (bf16 elems) for LDS tiles

static __device__ __forceinline__ unsigned bf16bits(float f) {
    union { float f; unsigned u; } x; x.f = f;
    return (x.u + 0x7FFFu + ((x.u >> 16) & 1u)) >> 16;   // round-to-nearest-even
}
static __device__ __forceinline__ __bf16 f2bf(float f) {
    unsigned short h = (unsigned short)bf16bits(f);
    return __builtin_bit_cast(__bf16, h);
}

// Async 16B global->LDS copy (CDNA5, tracked by ASYNCcnt). lds_off is the
// byte offset within the workgroup LDS allocation (= low 32 bits of the
// generic shared-aperture address); gptr is a global address.
static __device__ __forceinline__ void async_copy_b128(unsigned lds_off, const void* gptr) {
    asm volatile("global_load_async_to_lds_b128 %0, %1, off"
                 :: "v"(lds_off), "v"(gptr) : "memory");
}
static __device__ __forceinline__ void wait_async0() {
    asm volatile("s_wait_asynccnt 0x0" ::: "memory");
}

// ---------------------------------------------------------------------------
// Weight pre-pass: Win[K][N] f32  ->  Wt[N][K] bf16 (transposed, RNE)
// ---------------------------------------------------------------------------
__launch_bounds__(256)
__global__ void convert_transpose_kernel(const float* __restrict__ Win,
                                         __bf16* __restrict__ Wt, int K, int N)
{
    __shared__ float tile[32][33];
    int bk = blockIdx.y * 32, bn = blockIdx.x * 32;
    int tx = threadIdx.x & 31, ty = threadIdx.x >> 5;   // 32 x 8
    for (int i = 0; i < 32; i += 8) {
        int k = bk + ty + i, n = bn + tx;
        tile[ty + i][tx] = (k < K && n < N) ? Win[(size_t)k * N + n] : 0.f;
    }
    __syncthreads();
    for (int i = 0; i < 32; i += 8) {
        int n = bn + ty + i, k = bk + tx;
        if (n < N && k < K) Wt[(size_t)n * K + k] = f2bf(tile[tx][ty + i]);
    }
}

// ---------------------------------------------------------------------------
// Tiled GEMM: C[M,N] = epilogue(A[M,K] @ W[K,N] + bias)
//   A:  f32 in global, converted to bf16 while staging into LDS (packed b64)
//   Wt: pre-converted bf16, transposed [N][K]; staged via async global->LDS
//   WMMA bf16, f32 accumulate; fused bias + optional ReLU epilogue.
// ---------------------------------------------------------------------------
__launch_bounds__(256)
__global__ void gemm_bf16_wmma(const float* __restrict__ A,
                               const __bf16* __restrict__ Wt,
                               const float* __restrict__ bias,
                               float* __restrict__ C,
                               int M, int N, int K, int ldc, int relu)
{
    __shared__ __bf16 sA[BM * LDT];
    __shared__ __bf16 sB[BN * LDT];   // sB[n][k], k = 0..31

    const int tid  = threadIdx.x;
    const int lane = tid & 31;
    const int wave = tid >> 5;        // 0..7
    const int wm   = wave >> 1;       // 0..3 : 32-row strip
    const int wn   = wave & 1;        // 0..1 : 64-col strip

    const int block_m = blockIdx.y * BM;
    const int block_n = blockIdx.x * BN;

    const int lmod  = lane & 15;
    const int lhalf = (lane < 16) ? 0 : 1;

    const unsigned sB_off = (unsigned)(size_t)(void*)sB;   // LDS byte offset

    v8f acc[2][4];
    for (int i = 0; i < 2; ++i)
        for (int j = 0; j < 4; ++j)
            for (int e = 0; e < 8; ++e) acc[i][j][e] = 0.f;

    for (int k0 = 0; k0 < K; k0 += BK) {
        __syncthreads();

        // ---- B tile: 128 rows x 64B, async global->LDS, 2 x b128/thread ----
        for (int i = 0; i < 2; ++i) {
            int c  = tid * 2 + i;          // 0..511 (16B chunks)
            int n  = c >> 2;               // 0..127
            int k8 = (c & 3) * 8;          // bf16 elem offset in row
            int gn = block_n + n;
            if (gn < N) {
                const __bf16* src = Wt + (size_t)gn * K + k0 + k8;
                unsigned loff = sB_off + (unsigned)(n * LDT + k8) * 2u;
                async_copy_b128(loff, src);
            }
        }

        // ---- A tile (128x32 f32 -> bf16), packed pair stores (b64) ----
        {
            int r  = tid >> 3;             // 0..31
            int c4 = (tid & 7) * 4;        // 0,4,...,28
            for (int i = 0; i < 4; ++i) {
                int row  = r + i * 32;
                int grow = block_m + row;
                float4 v = make_float4(0.f, 0.f, 0.f, 0.f);
                if (grow < M)
                    v = *(const float4*)(A + (size_t)grow * K + k0 + c4);
                unsigned lo = bf16bits(v.x) | (bf16bits(v.y) << 16);
                unsigned hi = bf16bits(v.z) | (bf16bits(v.w) << 16);
                *(uint2*)&sA[row * LDT + c4] = make_uint2(lo, hi);
            }
        }

        wait_async0();
        __syncthreads();

        // ---- fragments per ISA 16-bit A/B layouts ----
        // A 16x32: lanes 0-15 hold K {0..7,16..23}, lanes 16-31 hold K {8..15,24..31}
        v16bf afrag[2];
        for (int i = 0; i < 2; ++i) {
            int row = wm * 32 + i * 16 + lmod;
            const __bf16* ap = &sA[row * LDT + lhalf * 8];
            v8bf lo = *(const v8bf*)ap;           // K = klo .. klo+7
            v8bf hi = *(const v8bf*)(ap + 16);    // K = klo+16 .. klo+23
            for (int e = 0; e < 8; ++e) { afrag[i][e] = lo[e]; afrag[i][8 + e] = hi[e]; }
        }
        // B 32x16: lanes 0-15 hold K {0..7,16..23} of col n, lanes 16-31 hold K {8..15,24..31}
        v16bf bfrag[4];
        for (int j = 0; j < 4; ++j) {
            int col = wn * 64 + j * 16 + lmod;
            const __bf16* bp = &sB[col * LDT + lhalf * 8];
            v8bf lo = *(const v8bf*)bp;
            v8bf hi = *(const v8bf*)(bp + 16);
            for (int e = 0; e < 8; ++e) { bfrag[j][e] = lo[e]; bfrag[j][8 + e] = hi[e]; }
        }

        for (int i = 0; i < 2; ++i)
            for (int j = 0; j < 4; ++j)
                acc[i][j] = __builtin_amdgcn_wmma_f32_16x16x32_bf16(
                    false, afrag[i], false, bfrag[j],
                    (short)0, acc[i][j], false, false);
    }

    // ---- epilogue: bias + optional ReLU; C/D layout: elem r -> M=r / M=8+r ----
    for (int i = 0; i < 2; ++i) {
        int mbase = block_m + wm * 32 + i * 16;
        for (int j = 0; j < 4; ++j) {
            int col = block_n + wn * 64 + j * 16 + lmod;
            if (col >= N) continue;
            float bv = bias ? bias[col] : 0.f;
            for (int r = 0; r < 8; ++r) {
                int row = mbase + ((lane < 16) ? r : 8 + r);
                if (row < M) {
                    float v = acc[i][j][r] + bv;
                    if (relu) v = fmaxf(v, 0.f);
                    C[(size_t)row * ldc + col] = v;
                }
            }
        }
    }
}

// ---------------------------------------------------------------------------
// LayerNorm: one wave per row; output row stride ldy (for concat into q_comb)
// ---------------------------------------------------------------------------
__launch_bounds__(256)
__global__ void layernorm_kernel(const float* __restrict__ X, float* __restrict__ Y,
                                 const float* __restrict__ g, const float* __restrict__ be,
                                 int rows, int ldy)
{
    int wave = threadIdx.x >> 5, lane = threadIdx.x & 31;
    int row = blockIdx.x * 8 + wave;
    if (row >= rows) return;
    const float* xr = X + (size_t)row * HDIM;
    float v[PER]; float s = 0.f;
    for (int i = 0; i < PER; ++i) { v[i] = xr[lane + 32 * i]; s += v[i]; }
    for (int m = 16; m; m >>= 1) s += __shfl_xor(s, m, 32);
    float mu = s * (1.f / HDIM);
    float q = 0.f;
    for (int i = 0; i < PER; ++i) { float d = v[i] - mu; q += d * d; }
    for (int m = 16; m; m >>= 1) q += __shfl_xor(q, m, 32);
    float rinv = rsqrtf(q * (1.f / HDIM) + 1e-5f);
    float* yr = Y + (size_t)row * ldy;
    for (int i = 0; i < PER; ++i) {
        int h = lane + 32 * i;
        yr[h] = g[h] * (v[i] - mu) * rinv + be[h];
    }
}

// ---------------------------------------------------------------------------
// Masked-mean prototypes: one block per support example (256 tokens)
// ---------------------------------------------------------------------------
__launch_bounds__(256)
__global__ void protos_kernel(const float* __restrict__ SA, const int* __restrict__ labels,
                              float* __restrict__ protos)
{
    __shared__ float mask[256];
    __shared__ float cnt_s;
    int b = blockIdx.x, t = threadIdx.x;
    mask[t] = (labels[b * 256 + t] > 0) ? 1.f : 0.f;
    __syncthreads();
    if (t == 0) {
        float c = 0.f;
        for (int s = 0; s < 256; ++s) c += mask[s];
        cnt_s = fmaxf(c, 1.f);
    }
    __syncthreads();
    float denom = cnt_s;
    const float* base = SA + (size_t)b * 256 * HDIM;
    for (int h = t; h < HDIM; h += 256) {
        float s = 0.f;
        for (int tok = 0; tok < 256; ++tok) s += mask[tok] * base[(size_t)tok * HDIM + h];
        protos[b * HDIM + h] = s / denom;
    }
}

// ---------------------------------------------------------------------------
// Prototype L2 normalization: one wave per prototype
// ---------------------------------------------------------------------------
__launch_bounds__(32)
__global__ void norm_protos_kernel(const float* __restrict__ protos, float* __restrict__ pn)
{
    int b = blockIdx.x, lane = threadIdx.x;
    const float* pr = protos + b * HDIM;
    float v[PER]; float ss = 0.f;
    for (int i = 0; i < PER; ++i) { v[i] = pr[lane + 32 * i]; ss += v[i] * v[i]; }
    for (int m = 16; m; m >>= 1) ss += __shfl_xor(ss, m, 32);
    float inv = 1.f / fmaxf(sqrtf(ss), 1e-8f);
    for (int i = 0; i < PER; ++i) pn[b * HDIM + lane + 32 * i] = v[i] * inv;
}

// ---------------------------------------------------------------------------
// Cosine-sim attention over 32 prototypes: one wave per query row
// (lane p == prototype p). Writes inp = x + softmax(qn.pn) @ protos
// ---------------------------------------------------------------------------
__launch_bounds__(256)
__global__ void attention_kernel(const float* __restrict__ X, const float* __restrict__ protos,
                                 const float* __restrict__ pn, float* __restrict__ inp, int rows)
{
    __shared__ float sx[8][HDIM];
    int wave = threadIdx.x >> 5, lane = threadIdx.x & 31;
    int row = blockIdx.x * 8 + wave;
    if (row >= rows) return;
    const float* xr = X + (size_t)row * HDIM;
    float xl[PER]; float ss = 0.f;
    for (int i = 0; i < PER; ++i) {
        float v = xr[lane + 32 * i];
        xl[i] = v;
        sx[wave][lane + 32 * i] = v;
        ss += v * v;
    }
    for (int m = 16; m; m >>= 1) ss += __shfl_xor(ss, m, 32);
    float inv = 1.f / fmaxf(sqrtf(ss), 1e-8f);

    // per-lane cosine similarity with prototype 'lane' (same-wave LDS is in order)
    const float* pr = pn + lane * HDIM;
    float sim = 0.f;
    for (int h = 0; h < HDIM; ++h) sim += sx[wave][h] * pr[h];
    sim *= inv;

    // softmax across the 32 lanes
    float mx = sim;
    for (int m = 16; m; m >>= 1) mx = fmaxf(mx, __shfl_xor(mx, m, 32));
    float e = expf(sim - mx);
    float se = e;
    for (int m = 16; m; m >>= 1) se += __shfl_xor(se, m, 32);
    float w = e / se;

    // infl = w @ protos ; out = x + infl
    float accv[PER];
    for (int i = 0; i < PER; ++i) accv[i] = xl[i];
    for (int p = 0; p < 32; ++p) {
        float wp = __shfl(w, p, 32);
        const float* prow = protos + p * HDIM;
        for (int i = 0; i < PER; ++i) accv[i] += wp * prow[lane + 32 * i];
    }
    float* outp = inp + (size_t)row * HDIM;
    for (int i = 0; i < PER; ++i) outp[lane + 32 * i] = accv[i];
}

// ---------------------------------------------------------------------------
// Final tiny layer: [rows,64] @ W3[64,3] + b3 -> preds[rows,3]
// ---------------------------------------------------------------------------
__launch_bounds__(256)
__global__ void mc3_kernel(const float* __restrict__ H2, const float* __restrict__ W3,
                           const float* __restrict__ b3, float* __restrict__ P, int rows)
{
    int r = blockIdx.x * 256 + threadIdx.x;
    if (r >= rows) return;
    const float* h = H2 + (size_t)r * 64;
    float a0 = b3[0], a1 = b3[1], a2 = b3[2];
    for (int k = 0; k < 64; ++k) {
        float hv = h[k];
        a0 += hv * W3[k * 3 + 0];
        a1 += hv * W3[k * 3 + 1];
        a2 += hv * W3[k * 3 + 2];
    }
    float* pr = P + (size_t)r * 3;
    pr[0] = a0; pr[1] = a1; pr[2] = a2;
}

// ---------------------------------------------------------------------------
// Host orchestration
// ---------------------------------------------------------------------------
extern "C" void kernel_launch(void* const* d_in, const int* in_sizes, int n_in,
                              void* d_out, int out_size, void* d_ws, size_t ws_size,
                              hipStream_t stream)
{
    (void)in_sizes; (void)n_in; (void)out_size; (void)ws_size;

    const float* sup   = (const float*)d_in[0];   // [32,256,768]
    const int*   lab   = (const int*)  d_in[1];   // [32,256]
    const float* qry   = (const float*)d_in[2];   // [64,512,768]
    const float* aw_W1 = (const float*)d_in[3];
    const float* aw_b1 = (const float*)d_in[4];
    const float* aw_W2 = (const float*)d_in[5];
    const float* aw_b2 = (const float*)d_in[6];
    const float* aw_g  = (const float*)d_in[7];
    const float* aw_be = (const float*)d_in[8];
    const float* ct_W1 = (const float*)d_in[9];
    const float* ct_b1 = (const float*)d_in[10];
    const float* ct_W2 = (const float*)d_in[11];
    const float* ct_b2 = (const float*)d_in[12];
    const float* ct_g  = (const float*)d_in[13];
    const float* ct_be = (const float*)d_in[14];
    const float* ad_W1 = (const float*)d_in[15];
    const float* ad_b1 = (const float*)d_in[16];
    const float* ad_W2 = (const float*)d_in[17];
    const float* ad_b2 = (const float*)d_in[18];
    const float* mc_W1 = (const float*)d_in[19];
    const float* mc_b1 = (const float*)d_in[20];
    const float* mc_W2 = (const float*)d_in[21];
    const float* mc_b2 = (const float*)d_in[22];
    const float* mc_W3 = (const float*)d_in[23];
    const float* mc_b3 = (const float*)d_in[24];

    const int Ms = 32 * 256;            // 8192 support rows
    const int Mq = 64 * 512;            // 32768 query rows
    const int CH = 8192;                // chunk rows
    const int NCH = Mq / CH;            // 4

    // output layout (flat f32, in tuple order): preds, adapted, protos, q_comb
    float* out     = (float*)d_out;
    float* preds   = out;                                   // 32768*3
    float* adapted = out + (size_t)Mq * 3;                  // 32768*768
    float* protos  = adapted + (size_t)Mq * HDIM;           // 32*768
    float* qcomb   = protos + (size_t)32 * HDIM;            // 32768*1536

    // workspace: ping-pong f32 buffers + pn + bf16 transposed weight pool
    float* bufA = (float*)d_ws;
    float* bufB = bufA + (size_t)CH * HDIM;
    float* pn   = bufB + (size_t)CH * HDIM;
    __bf16* wt_aw1 = (__bf16*)(pn + (size_t)32 * HDIM);     // [768][768]
    __bf16* wt_aw2 = wt_aw1 + (size_t)HDIM * HDIM;
    __bf16* wt_ct1 = wt_aw2 + (size_t)HDIM * HDIM;
    __bf16* wt_ct2 = wt_ct1 + (size_t)HDIM * HDIM;
    __bf16* wt_ad1 = wt_ct2 + (size_t)HDIM * HDIM;
    __bf16* wt_ad2 = wt_ad1 + (size_t)HDIM * HDIM;
    __bf16* wt_mc1 = wt_ad2 + (size_t)HDIM * HDIM;          // [768][1536]
    __bf16* wt_mc2 = wt_mc1 + (size_t)HDIM * 2 * HDIM;      // [64][768]

    dim3 blk(256);
    auto convT = [&](const float* W, __bf16* Wt, int K, int N) {
        dim3 grid((N + 31) / 32, (K + 31) / 32);
        hipLaunchKernelGGL(convert_transpose_kernel, grid, blk, 0, stream, W, Wt, K, N);
    };
    auto gemm = [&](const float* A, const __bf16* Wt, const float* b, float* C,
                    int M, int N, int K, int ldc, int relu) {
        dim3 grid((N + BN - 1) / BN, (M + BM - 1) / BM);
        hipLaunchKernelGGL(gemm_bf16_wmma, grid, blk, 0, stream, A, Wt, b, C, M, N, K, ldc, relu);
    };

    // ---- 0. pre-convert + transpose weights to bf16 (once per launch) ----
    convT(aw_W1, wt_aw1, HDIM, HDIM);
    convT(aw_W2, wt_aw2, HDIM, HDIM);
    convT(ct_W1, wt_ct1, HDIM, HDIM);
    convT(ct_W2, wt_ct2, HDIM, HDIM);
    convT(ad_W1, wt_ad1, HDIM, HDIM);
    convT(ad_W2, wt_ad2, HDIM, HDIM);
    convT(mc_W1, wt_mc1, 2 * HDIM, HDIM);
    convT(mc_W2, wt_mc2, HDIM, 64);

    // ---- 1. support aware-encoder (contr branch is dead code: skipped) ----
    gemm(sup,  wt_aw1, aw_b1, bufA, Ms, HDIM, HDIM, HDIM, 1);
    gemm(bufA, wt_aw2, aw_b2, bufB, Ms, HDIM, HDIM, HDIM, 0);
    hipLaunchKernelGGL(layernorm_kernel, dim3(Ms / 8), blk, 0, stream,
                       bufB, bufB, aw_g, aw_be, Ms, HDIM);
    hipLaunchKernelGGL(protos_kernel, dim3(32), blk, 0, stream, bufB, lab, protos);
    hipLaunchKernelGGL(norm_protos_kernel, dim3(32), dim3(32), 0, stream, protos, pn);

    // ---- 2. adaptation loop (x lives in d_out 'adapted' region) ----
    hipMemcpyAsync(adapted, qry, (size_t)Mq * HDIM * sizeof(float),
                   hipMemcpyDeviceToDevice, stream);
    for (int step = 0; step < 5; ++step) {
        for (int c = 0; c < NCH; ++c) {
            float* xc = adapted + (size_t)c * CH * HDIM;
            hipLaunchKernelGGL(attention_kernel, dim3(CH / 8), blk, 0, stream,
                               xc, protos, pn, bufA, CH);
            gemm(bufA, wt_ad1, ad_b1, bufB, CH, HDIM, HDIM, HDIM, 1);
            gemm(bufB, wt_ad2, ad_b2, xc,   CH, HDIM, HDIM, HDIM, 0);
        }
    }

    // ---- 3. dual heads + meta-classifier, per chunk ----
    for (int c = 0; c < NCH; ++c) {
        float* xc = adapted + (size_t)c * CH * HDIM;
        float* qc = qcomb   + (size_t)c * CH * (2 * HDIM);

        gemm(xc,   wt_aw1, aw_b1, bufA, CH, HDIM, HDIM, HDIM, 1);
        gemm(bufA, wt_aw2, aw_b2, bufB, CH, HDIM, HDIM, HDIM, 0);
        hipLaunchKernelGGL(layernorm_kernel, dim3(CH / 8), blk, 0, stream,
                           bufB, qc, aw_g, aw_be, CH, 2 * HDIM);

        gemm(xc,   wt_ct1, ct_b1, bufA, CH, HDIM, HDIM, HDIM, 1);
        gemm(bufA, wt_ct2, ct_b2, bufB, CH, HDIM, HDIM, HDIM, 0);
        hipLaunchKernelGGL(layernorm_kernel, dim3(CH / 8), blk, 0, stream,
                           bufB, qc + HDIM, ct_g, ct_be, CH, 2 * HDIM);

        gemm(qc,   wt_mc1, mc_b1, bufA, CH, HDIM, 2 * HDIM, HDIM, 1);
        gemm(bufA, wt_mc2, mc_b2, bufB, CH, 64,   HDIM,     64,   1);
        hipLaunchKernelGGL(mc3_kernel, dim3(CH / 256), blk, 0, stream,
                           bufB, mc_W3, mc_b3, preds + (size_t)c * CH * 3, CH);
    }
}
